// RMAC_87411174408226
// MI455X (gfx1250) — compile-verified, matrix operand order checked by reading, not run on hardware
//
#include <hip/hip_runtime.h>
#include <hip/hip_bf16.h>
#include <stdint.h>

// ---------------------------------------------------------------------------
// R-MAC pooling for fea (64, 2048, 14, 14) f32 on MI455X (gfx1250).
//
// Memory-bound: 102.8 MB single-pass read -> ~4.4 us floor at 23.3 TB/s.
// Kernel 1 streams planes through LDS with CDNA5 async global->LDS copies.
// Kernel 2 does the per-batch channel norm with v_wmma_f32_16x16x32_f16
// (ones-matrix reduction with exact hi/lo f16 splitting) and combines.
// ---------------------------------------------------------------------------

typedef __attribute__((ext_vector_type(16))) _Float16 v16h;
typedef __attribute__((ext_vector_type(8)))  _Float16 v8h;
typedef __attribute__((ext_vector_type(8)))  float    v8f;

#define NPLANES   (64 * 2048)     // total (batch, channel) planes
#define PLANE_ELM 196             // 14*14
#define PLANE_B   784             // bytes per plane
#define K1_THR    128             // planes staged per block (4 waves)
#define K1_STAGE_B (K1_THR * PLANE_B)  // 100352 B of dynamic LDS

// ---------------------------------------------------------------------------
// Kernel 1: stage 128 planes into LDS via async copies, compute 14 region
// maxes per plane with a separable row/column max scheme.
// Regions (start,row/col width actually used = width-1):
//   r0          : rows 0..12,  cols 0..12           (13x13)
//   r1..r4      : rows {0,2}+0..7, cols {0,2}+0..7  (8x8)
//   r5..r13     : rows {0,2,4}+0..5, cols {0,2,4}+0..5 (6x6)
// ---------------------------------------------------------------------------
__global__ __launch_bounds__(K1_THR) void rmac_regionmax(
    const float* __restrict__ fea, float* __restrict__ ws) {
  extern __shared__ char smem[];
  const int tid = threadIdx.x;
  const long long blockPlane = (long long)blockIdx.x * K1_THR;
  const float* src = fea + blockPlane * PLANE_ELM;  // 784B*blk -> 16B aligned

  // Low 32 bits of the flat shared address are the wave-relative LDS offset.
  unsigned ldsBase = (unsigned)(uintptr_t)smem;

  // Coalesced async staging: 49 iterations x (128 lanes x 16B) = 100352 B.
  // global_load_async_to_lds_b128 vdst(lds addr), vaddr(u32 off), saddr(base)
#pragma unroll
  for (int it = 0; it < 49; ++it) {
    unsigned off = (unsigned)(it * (K1_THR * 16) + tid * 16);
    unsigned lds = ldsBase + off;
    asm volatile("global_load_async_to_lds_b128 %0, %1, %2"
                 :: "v"(lds), "v"(off), "s"(src) : "memory");
  }
  asm volatile("s_wait_asynccnt 0" ::: "memory");
  __syncthreads();

  // Each lane reduces its own plane (784 B contiguous in LDS).
  const char* myPlane = smem + tid * PLANE_B;
  float r[14];
#pragma unroll
  for (int i = 0; i < 14; ++i) r[i] = -3.402823466e38f;

#pragma unroll
  for (int h = 0; h < 14; ++h) {
    const float2* row = (const float2*)(myPlane + h * 56);
    float2 p0 = row[0], p1 = row[1], p2 = row[2], p3 = row[3];
    float2 p4 = row[4], p5 = row[5], p6 = row[6];
    float m01 = fmaxf(p0.x, p0.y), m23 = fmaxf(p1.x, p1.y);
    float m45 = fmaxf(p2.x, p2.y), m67 = fmaxf(p3.x, p3.y);
    float m89 = fmaxf(p4.x, p4.y), mAB = fmaxf(p5.x, p5.y);
    // column-window maxes for this row
    float c6_0 = fmaxf(fmaxf(m01, m23), m45);   // cols 0..5
    float c6_2 = fmaxf(fmaxf(m23, m45), m67);   // cols 2..7
    float c6_4 = fmaxf(fmaxf(m45, m67), m89);   // cols 4..9
    float c8_0 = fmaxf(c6_0, m67);              // cols 0..7
    float c8_2 = fmaxf(c6_2, m89);              // cols 2..9
    float c13  = fmaxf(fmaxf(c8_0, m89), fmaxf(mAB, p6.x)); // cols 0..12
    // fold into region accumulators (h is compile-time after unroll)
    if (h <= 12)            r[0]  = fmaxf(r[0],  c13);
    if (h <= 7) {           r[1]  = fmaxf(r[1],  c8_0);
                            r[2]  = fmaxf(r[2],  c8_2); }
    if (h >= 2 && h <= 9) { r[3]  = fmaxf(r[3],  c8_0);
                            r[4]  = fmaxf(r[4],  c8_2); }
    if (h <= 5) {           r[5]  = fmaxf(r[5],  c6_0);
                            r[6]  = fmaxf(r[6],  c6_2);
                            r[7]  = fmaxf(r[7],  c6_4); }
    if (h >= 2 && h <= 7) { r[8]  = fmaxf(r[8],  c6_0);
                            r[9]  = fmaxf(r[9],  c6_2);
                            r[10] = fmaxf(r[10], c6_4); }
    if (h >= 4 && h <= 9) { r[11] = fmaxf(r[11], c6_0);
                            r[12] = fmaxf(r[12], c6_2);
                            r[13] = fmaxf(r[13], c6_4); }
  }

  const long long plane = blockPlane + tid;
#pragma unroll
  for (int i = 0; i < 14; ++i)
    ws[(long long)i * NPLANES + plane] = r[i];  // coalesced per i
}

// ---------------------------------------------------------------------------
// Kernel 2: one block per batch. Per-thread partial sum-of-squares over 8
// channels for all 14 regions; block reduction via v_wmma_f32_16x16x32_f16
// against an all-ones B matrix, with hi/lo f16 splitting of each partial so
// the f32 WMMA accumulator reproduces the f32 sum almost exactly.
// Then out[b][c] = sum_r ws[r][b][c] * rsqrt(normsq[r]).
// ---------------------------------------------------------------------------
__global__ __launch_bounds__(256) void rmac_norm_combine(
    const float* __restrict__ ws, float* __restrict__ out) {
  const int b = blockIdx.x, t = threadIdx.x;
  const int base = b * 2048;

  float ss[14];
#pragma unroll
  for (int rr = 0; rr < 14; ++rr) ss[rr] = 0.f;
#pragma unroll
  for (int k = 0; k < 8; ++k) {
    const int idx = base + t + 256 * k;
#pragma unroll
    for (int rr = 0; rr < 14; ++rr) {
      float v = ws[rr * NPLANES + idx];
      ss[rr] = fmaf(v, v, ss[rr]);
    }
  }

  __shared__ _Float16 hiv[16][256];   // rows 14,15 zero-padded
  __shared__ _Float16 lov[16][256];
  __shared__ float    invn[16];
#pragma unroll
  for (int rr = 0; rr < 14; ++rr) {
    float s = ss[rr];
    _Float16 hh = (_Float16)s;
    hiv[rr][t] = hh;
    lov[rr][t] = (_Float16)(s - (float)hh);
  }
  hiv[14][t] = (_Float16)0.f; hiv[15][t] = (_Float16)0.f;
  lov[14][t] = (_Float16)0.f; lov[15][t] = (_Float16)0.f;
  __syncthreads();

  if (t < 32) {  // wave 0, EXEC all-1s as WMMA requires
    v16h ones;
#pragma unroll
    for (int i = 0; i < 16; ++i) ones[i] = (_Float16)1.0f;
    v8f acc = {};
    const int M   = t & 15;        // A-matrix row this lane feeds
    const int sel = (t >> 4) & 1;  // lane 16..31 supplies K=8..15 / 24..31
#pragma unroll
    for (int chunk = 0; chunk < 8; ++chunk) {
      const int c0 = chunk * 32 + sel * 8;       // A[M][K=0..7 | 8..15]
      const int c1 = chunk * 32 + 16 + sel * 8;  // A[M][K=16..23 | 24..31]
      v8h ha = *(const v8h*)&hiv[M][c0];
      v8h hb = *(const v8h*)&hiv[M][c1];
      v8h la = *(const v8h*)&lov[M][c0];
      v8h lb = *(const v8h*)&lov[M][c1];
      v16h Ah = __builtin_shufflevector(ha, hb, 0,1,2,3,4,5,6,7,8,9,10,11,12,13,14,15);
      v16h Al = __builtin_shufflevector(la, lb, 0,1,2,3,4,5,6,7,8,9,10,11,12,13,14,15);
      acc = __builtin_amdgcn_wmma_f32_16x16x32_f16(false, Ah, false, ones,
                                                   (short)0, acc, false, false);
      acc = __builtin_amdgcn_wmma_f32_16x16x32_f16(false, Al, false, ones,
                                                   (short)0, acc, false, false);
    }
    // D layout: VGPR j, lanes 0-15 -> row j (N=lane); lanes 16-31 -> row j+8.
    if (t == 0) {
#pragma unroll
      for (int j = 0; j < 8; ++j) invn[j] = 1.0f / sqrtf(acc[j]);
    }
    if (t == 16) {
#pragma unroll
      for (int j = 0; j < 6; ++j) invn[8 + j] = 1.0f / sqrtf(acc[j]);
    }
  }
  __syncthreads();

  float w[14];
#pragma unroll
  for (int rr = 0; rr < 14; ++rr) w[rr] = invn[rr];
#pragma unroll
  for (int k = 0; k < 8; ++k) {
    const int idx = base + t + 256 * k;
    float o = 0.f;
#pragma unroll
    for (int rr = 0; rr < 14; ++rr)
      o = fmaf(ws[rr * NPLANES + idx], w[rr], o);  // ws is L2-hot
    out[idx] = o;
  }
}

// ---------------------------------------------------------------------------
extern "C" void kernel_launch(void* const* d_in, const int* in_sizes, int n_in,
                              void* d_out, int out_size, void* d_ws, size_t ws_size,
                              hipStream_t stream) {
  (void)in_sizes; (void)n_in; (void)out_size; (void)ws_size;
  const float* fea = (const float*)d_in[0];
  float*       out = (float*)d_out;
  float*       ws  = (float*)d_ws;   // needs 14 * 131072 * 4 = 7.34 MB

  dim3 g1(NPLANES / K1_THR), b1(K1_THR);
  rmac_regionmax<<<g1, b1, K1_STAGE_B, stream>>>(fea, ws);
  rmac_norm_combine<<<64, 256, 0, stream>>>(ws, out);
}
